// MAB_81123342287550
// MI455X (gfx1250) — matrix operand order repaired and checked by reading, not compiled, not run on hardware
//
#include <hip/hip_runtime.h>
#include <cstdint>

// ---- types -----------------------------------------------------------------
typedef __attribute__((ext_vector_type(16))) __bf16 v16bf;
typedef __attribute__((ext_vector_type(8)))  __bf16 bf16x8;
typedef __attribute__((ext_vector_type(4)))  __bf16 bf16x4;
typedef __attribute__((ext_vector_type(8)))  float  v8f;

#define NHEAD 8
#define DMODEL 1024
#define DHEAD 128
#define NSEQ 2048
#define NBATCH 4
#define LN_EPS 1e-5f

// ---- CDNA5 async global->LDS copy (ASYNCcnt-tracked, ISA §15.18.3) ---------
// Low 32 bits of a flat pointer into LDS are the LDS byte address.
__device__ __forceinline__ void async_g2l_b128(const void* g, void* lds) {
  unsigned l = (unsigned)(uintptr_t)lds;
  asm volatile("global_load_async_to_lds_b128 %0, %1, off"
               :: "v"(l), "v"(g) : "memory");
}
__device__ __forceinline__ void wait_async0() {
  asm volatile("s_wait_asynccnt 0x0" ::: "memory");
}

// ============================================================================
// GEMM: Y = epilogue(X @ W^T + bias), X:[M,1024] f32, W:[1024,1024] f32 row-major
// MODE 0: Y f32 row-major
// MODE 1: Y bf16 head-major rows   [b][h][row][dh]      (V projection)
// MODE 3: Y bf16 head-major transposed [b][h][dh][row]  (K projection -> K^T)
// MODE 2: Y f32 = RES + relu(X@W^T + bias)              (output proj + residual)
// Block: 256 threads (8 waves), 128x128 tile; wave w owns rows [16w,16w+16).
// ============================================================================
template <int MODE>
__global__ __launch_bounds__(256) void gemm_bias_kernel(
    const float* __restrict__ X, const float* __restrict__ W,
    const float* __restrict__ bias, const float* __restrict__ RES,
    float* __restrict__ Yf, __bf16* __restrict__ Yh, int Mrows) {
  __shared__ __bf16 Asm[128][32];   // X tile, bf16
  __shared__ __bf16 Bsm[32][128];   // W tile transposed: Bsm[k][n]

  const int tid  = threadIdx.x;
  const int wave = tid >> 5;
  const int lane = tid & 31;
  const int half = lane >> 4;
  const int m0   = blockIdx.y * 128;
  const int n0   = blockIdx.x * 128;

  v8f acc[8] = {};

  for (int k0 = 0; k0 < DMODEL; k0 += 32) {
    if (k0 + 32 < DMODEL) {  // emits global_prefetch_b8
      __builtin_prefetch(X + (size_t)(m0 + (tid >> 1)) * DMODEL + k0 + 32, 0, 1);
      __builtin_prefetch(W + (size_t)(n0 + (tid >> 1)) * DMODEL + k0 + 32, 0, 1);
    }
    // stage A: 128x32 fp32 -> bf16 (contiguous)
#pragma unroll
    for (int j = 0; j < 4; ++j) {
      int e4 = tid * 4 + j;        // 1024 float4 chunks
      int r  = e4 >> 3;            // 8 float4 per 32-wide row
      int c  = (e4 & 7) * 4;
      float4 f = *(const float4*)(X + (size_t)(m0 + r) * DMODEL + k0 + c);
      bf16x4 hv;
      hv.x = (__bf16)f.x; hv.y = (__bf16)f.y; hv.z = (__bf16)f.z; hv.w = (__bf16)f.w;
      *(bf16x4*)(&Asm[r][c]) = hv;
    }
    // stage B transposed: W rows are contiguous along k; scatter into Bsm[k][n]
#pragma unroll
    for (int j = 0; j < 4; ++j) {
      int e4 = tid * 4 + j;
      int n  = e4 >> 3;
      int c  = (e4 & 7) * 4;
      float4 f = *(const float4*)(W + (size_t)(n0 + n) * DMODEL + k0 + c);
      Bsm[c + 0][n] = (__bf16)f.x;
      Bsm[c + 1][n] = (__bf16)f.y;
      Bsm[c + 2][n] = (__bf16)f.z;
      Bsm[c + 3][n] = (__bf16)f.w;
    }
    __syncthreads();

    // A fragment (ISA layout: lane<16 -> K{0..7,16..23}, lane>=16 -> K{8..15,24..31})
    const int row = wave * 16 + (lane & 15);
    v16bf afrag;
    {
      bf16x8 lo = *(const bf16x8*)(&Asm[row][half * 8]);
      bf16x8 hi = *(const bf16x8*)(&Asm[row][half * 8 + 16]);
#pragma unroll
      for (int i = 0; i < 8; ++i) { afrag[i] = lo[i]; afrag[i + 8] = hi[i]; }
    }
#pragma unroll
    for (int nt = 0; nt < 8; ++nt) {
      v16bf bfrag;
#pragma unroll
      for (int i = 0; i < 8; ++i) {
        bfrag[i]     = Bsm[half * 8 + i][nt * 16 + (lane & 15)];
        bfrag[i + 8] = Bsm[16 + half * 8 + i][nt * 16 + (lane & 15)];
      }
      acc[nt] = __builtin_amdgcn_wmma_f32_16x16x32_bf16(
          false, afrag, false, bfrag, (short)0, acc[nt], false, false);
    }
    __syncthreads();
  }

  // epilogue (C layout: VGPR i -> row i (+8 for upper half-lanes), lane&15 -> col)
#pragma unroll
  for (int nt = 0; nt < 8; ++nt) {
#pragma unroll
    for (int i = 0; i < 8; ++i) {
      int r = m0 + wave * 16 + half * 8 + i;
      int c = n0 + nt * 16 + (lane & 15);
      float v = acc[nt][i] + bias[c];
      if (MODE == 0) {
        Yf[(size_t)r * DMODEL + c] = v;
      } else if (MODE == 1) {
        int bb = r >> 11, rr = r & (NSEQ - 1);
        int hh = c >> 7,  dd = c & (DHEAD - 1);
        Yh[(((size_t)(bb * NHEAD + hh)) * NSEQ + rr) * DHEAD + dd] = (__bf16)v;
      } else if (MODE == 3) {
        int bb = r >> 11, rr = r & (NSEQ - 1);
        int hh = c >> 7,  dd = c & (DHEAD - 1);
        Yh[(((size_t)(bb * NHEAD + hh)) * DHEAD + dd) * NSEQ + rr] = (__bf16)v;
      } else {
        float res = RES[(size_t)r * DMODEL + c];
        Yf[(size_t)r * DMODEL + c] = res + fmaxf(v, 0.0f);
      }
    }
  }
}

// ============================================================================
// Flash attention per (128 q-rows, head, batch).  o = q + softmax(qk^T/sqrt(d)) v
// K^T and V tiles stream through GLOBAL_LOAD_ASYNC_TO_LDS_B128 (ASYNCcnt).
// ============================================================================
__global__ __launch_bounds__(256) void attn_kernel(
    const float* __restrict__ qf, const __bf16* __restrict__ khT,
    const __bf16* __restrict__ vh, float* __restrict__ O) {
  __shared__ __bf16 QP[128][DHEAD];   // Q tile (bf16, prescaled); reused as P
  __shared__ __bf16 Kst[DHEAD][128];  // K^T tile: Kst[d][kr]
  __shared__ __bf16 Vs[128][DHEAD];   // V tile (row-major)

  const int tid  = threadIdx.x;
  const int wave = tid >> 5;
  const int lane = tid & 31;
  const int half = lane >> 4;
  const int qt   = blockIdx.x;
  const int head = blockIdx.y;
  const int bat  = blockIdx.z;
  const float scale = 0.08838834764831845f;  // 1/sqrt(128)

  // ---- load Q tile (fp32 -> bf16, prescaled; needs conversion -> VGPR path) -
  const float* qbase = qf + ((size_t)(bat * NSEQ + qt * 128)) * DMODEL + head * DHEAD;
#pragma unroll
  for (int j = 0; j < 16; ++j) {
    int e4 = tid * 16 + j;       // 4096 float4 chunks
    int r  = e4 >> 5;            // 32 float4 per 128-wide row
    int c  = (e4 & 31) * 4;
    float4 f = *(const float4*)(qbase + (size_t)r * DMODEL + c);
    bf16x4 hv;
    hv.x = (__bf16)(f.x * scale); hv.y = (__bf16)(f.y * scale);
    hv.z = (__bf16)(f.z * scale); hv.w = (__bf16)(f.w * scale);
    *(bf16x4*)(&QP[r][c]) = hv;
  }
  __syncthreads();

  // ---- preload Q A-fragments (d=128 -> 4 K-steps) --------------------------
  const int qrow = wave * 16 + (lane & 15);
  v16bf aq[4];
#pragma unroll
  for (int kk = 0; kk < 4; ++kk) {
    const __bf16* base = &QP[qrow][kk * 32];
    bf16x8 lo = *(const bf16x8*)(base + half * 8);
    bf16x8 hi = *(const bf16x8*)(base + half * 8 + 16);
#pragma unroll
    for (int i = 0; i < 8; ++i) { aq[kk][i] = lo[i]; aq[kk][i + 8] = hi[i]; }
  }

  v8f oacc[8] = {};
  float mrow[8], lrow[8];
#pragma unroll
  for (int i = 0; i < 8; ++i) { mrow[i] = -1e30f; lrow[i] = 0.0f; }

  const __bf16* kbase = khT + ((size_t)(bat * NHEAD + head)) * DHEAD * NSEQ;  // [d][seq]
  const __bf16* vbase = vh  + ((size_t)(bat * NHEAD + head)) * NSEQ * DHEAD;  // [seq][d]

  for (int kb = 0; kb < NSEQ; kb += 128) {
    // ---- async-stage K^T tile (strided rows) and V tile (contiguous 32 KB) -
#pragma unroll
    for (int j = 0; j < 8; ++j) {
      int e = j * 256 + tid;         // 2048 chunks of 16 B
      int r = e >> 4;                // d row
      int c = (e & 15) * 8;          // bf16 column
      async_g2l_b128(kbase + (size_t)r * NSEQ + kb + c, &Kst[r][c]);
    }
    {
      const __bf16* vsrc = vbase + (size_t)kb * DHEAD;
      __bf16* vdst = &Vs[0][0];
#pragma unroll
      for (int j = 0; j < 8; ++j) {
        int e = j * 256 + tid;
        async_g2l_b128(vsrc + (size_t)e * 8, vdst + (size_t)e * 8);
      }
    }
    wait_async0();
    __syncthreads();

    // ---- S = Q @ K^T (16 rows x 128 cols per wave) -------------------------
    v8f s[8] = {};
#pragma unroll
    for (int kk = 0; kk < 4; ++kk) {
#pragma unroll
      for (int nt = 0; nt < 8; ++nt) {
        v16bf bfrag;
#pragma unroll
        for (int i = 0; i < 8; ++i) {
          bfrag[i]     = Kst[kk * 32 + half * 8 + i][nt * 16 + (lane & 15)];
          bfrag[i + 8] = Kst[kk * 32 + 16 + half * 8 + i][nt * 16 + (lane & 15)];
        }
        s[nt] = __builtin_amdgcn_wmma_f32_16x16x32_bf16(
            false, aq[kk], false, bfrag, (short)0, s[nt], false, false);
      }
    }

    // ---- online softmax (row = 16 lanes of one VGPR slot) ------------------
#pragma unroll
    for (int i = 0; i < 8; ++i) {
      float mx = s[0][i];
#pragma unroll
      for (int nt = 1; nt < 8; ++nt) mx = fmaxf(mx, s[nt][i]);
      mx = fmaxf(mx, __shfl_xor(mx, 1, 32));
      mx = fmaxf(mx, __shfl_xor(mx, 2, 32));
      mx = fmaxf(mx, __shfl_xor(mx, 4, 32));
      mx = fmaxf(mx, __shfl_xor(mx, 8, 32));
      float mnew  = fmaxf(mrow[i], mx);
      float alpha = __expf(mrow[i] - mnew);
      float rsum = 0.0f;
#pragma unroll
      for (int nt = 0; nt < 8; ++nt) {
        float p = __expf(s[nt][i] - mnew);
        s[nt][i] = p;
        rsum += p;
      }
      rsum += __shfl_xor(rsum, 1, 32);
      rsum += __shfl_xor(rsum, 2, 32);
      rsum += __shfl_xor(rsum, 4, 32);
      rsum += __shfl_xor(rsum, 8, 32);
      lrow[i] = lrow[i] * alpha + rsum;
      mrow[i] = mnew;
#pragma unroll
      for (int nt = 0; nt < 8; ++nt) oacc[nt][i] *= alpha;
    }

    // ---- P: C-layout -> A-layout via (wave-private) LDS round trip ---------
    __bf16* Ps = &QP[0][0];  // reuse Q tile space; wave touches only its rows
#pragma unroll
    for (int nt = 0; nt < 8; ++nt)
#pragma unroll
      for (int i = 0; i < 8; ++i)
        Ps[(wave * 16 + half * 8 + i) * 128 + nt * 16 + (lane & 15)] = (__bf16)s[nt][i];

    // ---- O += P @ V --------------------------------------------------------
#pragma unroll
    for (int kk = 0; kk < 4; ++kk) {
      v16bf ap;
      const __bf16* base = Ps + qrow * 128 + kk * 32;
      bf16x8 lo = *(const bf16x8*)(base + half * 8);
      bf16x8 hi = *(const bf16x8*)(base + half * 8 + 16);
#pragma unroll
      for (int i = 0; i < 8; ++i) { ap[i] = lo[i]; ap[i + 8] = hi[i]; }
#pragma unroll
      for (int nt = 0; nt < 8; ++nt) {
        v16bf bfrag;
#pragma unroll
        for (int i = 0; i < 8; ++i) {
          bfrag[i]     = Vs[kk * 32 + half * 8 + i][nt * 16 + (lane & 15)];
          bfrag[i + 8] = Vs[kk * 32 + 16 + half * 8 + i][nt * 16 + (lane & 15)];
        }
        oacc[nt] = __builtin_amdgcn_wmma_f32_16x16x32_bf16(
            false, ap, false, bfrag, (short)0, oacc[nt], false, false);
      }
    }
    __syncthreads();
  }

  // ---- epilogue: o = q + O/l ----------------------------------------------
  float* obase = O + ((size_t)(bat * NSEQ + qt * 128)) * DMODEL + head * DHEAD;
#pragma unroll
  for (int nt = 0; nt < 8; ++nt) {
#pragma unroll
    for (int i = 0; i < 8; ++i) {
      int r = wave * 16 + half * 8 + i;
      int c = nt * 16 + (lane & 15);
      float invl = 1.0f / lrow[i];
      obase[(size_t)r * DMODEL + c] =
          qbase[(size_t)r * DMODEL + c] + oacc[nt][i] * invl;
    }
  }
}

// ============================================================================
// LayerNorm over 1024-wide rows; one wave per row, 8 rows per block.
// ============================================================================
__global__ __launch_bounds__(256) void ln_kernel(
    const float* __restrict__ X, const float* __restrict__ g,
    const float* __restrict__ bta, float* __restrict__ Y) {
  const int wave = threadIdx.x >> 5;
  const int lane = threadIdx.x & 31;
  const size_t row = (size_t)blockIdx.x * 8 + wave;
  const float* xr = X + row * DMODEL;

  float vals[32];
  float s = 0.0f;
#pragma unroll
  for (int j = 0; j < 32; ++j) { vals[j] = xr[j * 32 + lane]; s += vals[j]; }
#pragma unroll
  for (int m = 1; m < 32; m <<= 1) s += __shfl_xor(s, m, 32);
  float mean = s * (1.0f / 1024.0f);

  float vs = 0.0f;
#pragma unroll
  for (int j = 0; j < 32; ++j) { float d = vals[j] - mean; vs += d * d; }
#pragma unroll
  for (int m = 1; m < 32; m <<= 1) vs += __shfl_xor(vs, m, 32);
  float rstd = rsqrtf(vs * (1.0f / 1024.0f) + LN_EPS);

  float* yr = Y + row * DMODEL;
#pragma unroll
  for (int j = 0; j < 32; ++j) {
    int c = j * 32 + lane;
    yr[c] = (vals[j] - mean) * rstd * g[c] + bta[c];
  }
}

// ============================================================================
// launcher
// ============================================================================
extern "C" void kernel_launch(void* const* d_in, const int* in_sizes, int n_in,
                              void* d_out, int out_size, void* d_ws, size_t ws_size,
                              hipStream_t stream) {
  const float* Q  = (const float*)d_in[0];
  const float* K  = (const float*)d_in[1];
  const float* Wq = (const float*)d_in[2];
  const float* bq = (const float*)d_in[3];
  const float* Wk = (const float*)d_in[4];
  const float* bk = (const float*)d_in[5];
  const float* Wv = (const float*)d_in[6];
  const float* bv = (const float*)d_in[7];
  const float* Wo = (const float*)d_in[8];
  const float* bo = (const float*)d_in[9];
  const float* g0 = (const float*)d_in[10];
  const float* b0 = (const float*)d_in[11];
  const float* g1 = (const float*)d_in[12];
  const float* b1 = (const float*)d_in[13];
  float* out = (float*)d_out;

  char* ws = (char*)d_ws;
  float*  qf   = (float*)(ws);                         // 32 MB  q (fp32, residual)
  __bf16* khT  = (__bf16*)(ws + ((size_t)32 << 20));   // 16 MB  K^T head-major bf16
  __bf16* vhB  = (__bf16*)(ws + ((size_t)48 << 20));   // 16 MB  V head-major bf16
  float*  Obuf = (float*)(ws + ((size_t)64 << 20));    // 32 MB  attention out
  float*  X1   = (float*)(ws + ((size_t)96 << 20));    // 32 MB  LN0 out
  float*  Y2   = (float*)(ws + ((size_t)128 << 20));   // 32 MB  proj+residual out

  const int M = NBATCH * NSEQ;  // 8192
  dim3 blk(256);
  dim3 ggemm(DMODEL / 128, M / 128);  // (8, 64)

  gemm_bias_kernel<0><<<ggemm, blk, 0, stream>>>(Q, Wq, bq, nullptr, qf, nullptr, M);
  gemm_bias_kernel<3><<<ggemm, blk, 0, stream>>>(K, Wk, bk, nullptr, nullptr, khT, M);
  gemm_bias_kernel<1><<<ggemm, blk, 0, stream>>>(K, Wv, bv, nullptr, nullptr, vhB, M);
  attn_kernel<<<dim3(NSEQ / 128, NHEAD, NBATCH), blk, 0, stream>>>(qf, khT, vhB, Obuf);
  ln_kernel<<<dim3(M / 8), blk, 0, stream>>>(Obuf, g0, b0, X1);
  gemm_bias_kernel<2><<<ggemm, blk, 0, stream>>>(X1, Wo, bo, X1, Y2, nullptr, M);
  ln_kernel<<<dim3(M / 8), blk, 0, stream>>>(Y2, g1, b1, out);
}
